// GCN_30219389894771
// MI455X (gfx1250) — compile-verified, matrix operand order checked by reading, not run on hardware
//
#include <hip/hip_runtime.h>
#include <hip/hip_bf16.h>
#include <cstdint>
#include <cstddef>

// ---------------------------------------------------------------------------
// MI455X / gfx1250 implementation. All large GEMMs run on the wave32 WMMA
// path (v_wmma_f32_16x16x32_f16), activations stored f16, accumulation f32.
// B (weight) tiles are staged in LDS and shared by all 8 waves of the block;
// each wave owns a 32(M) x NJ*16(N) tile so every B fragment feeds 2 WMMAs.
// ---------------------------------------------------------------------------

typedef __attribute__((ext_vector_type(16))) _Float16 v16h;
typedef __attribute__((ext_vector_type(8)))  float    v8f;
typedef _Float16 half_t;

#define ACT_NONE 0
#define ACT_SILU 1
#define ACT_RELU 2
#define ACT_TANH 3

__device__ __forceinline__ float actf(float v, int a) {
  if (a == ACT_SILU) return v / (1.f + __expf(-v));
  if (a == ACT_RELU) return v > 0.f ? v : 0.f;
  if (a == ACT_TANH) return tanhf(v);
  return v;
}

union FragU { uint4 u[2]; v16h h; };
union H4   { uint2 u; half_t h[4]; };

#define LDS_CHUNK  128          // K halves staged per chunk
#define LDS_STRIDE 136          // padded row stride (halves) -> bank spread

// ---------------------------------------------------------------------------
// WMMA GEMM: C[M,N] = act(A[M,K] * B[K,N] + bias) (+ residual)
// A: f16 row-major, lda = Kp (mult of 32). B supplied transposed as
// BT[Np,Kp] f16 (row n holds column n of B). One wave computes a 32 x NJ*16
// tile (two M-tiles share each B fragment); the block's 64-column B chunk is
// staged in LDS and shared by all 8 waves. A fragment per ISA 16-bit A
// layout: lane l (m = l&15, h = l>>4) holds K = [8h..8h+7] and
// [16+8h..16+8h+7] -> two b128 loads. B fragment mirrors A with N in place
// of M. C layout: lane l col n = l&15, VGPR r row m = r + 8*(l>>4).
// NJ is compile-time: no branches around WMMA.
// ---------------------------------------------------------------------------
template<int NJ>
__global__ __launch_bounds__(256) void k_wmma_gemm(
    const half_t* __restrict__ A, int lda,
    const half_t* __restrict__ BT, int ldb,
    int Mtiles, int n0base, int Kp,
    const float* __restrict__ bias, int Nvalid, int act,
    const float* resid, int ldr,
    float* Cf, int ldcf,
    half_t* Ch, int ldch)
{
  __shared__ half_t lds[64 * LDS_STRIDE];

  const int wid  = threadIdx.x >> 5;
  const int lane = threadIdx.x & 31;
  const int tmb  = (blockIdx.x * 8 + wid) * 2;   // two M tiles per wave
  const bool wr0 = (tmb     < Mtiles);
  const bool wr1 = (tmb + 1 < Mtiles);
  const int tm0  = wr0 ? tmb       : (Mtiles - 1);  // clamp: stay alive for barriers
  const int tm1  = wr1 ? (tmb + 1) : (Mtiles - 1);
  const int n0   = n0base + blockIdx.y * 64;
  const int l15  = lane & 15;
  const int hlf  = lane >> 4;

  v8f acc0[NJ], acc1[NJ];
#pragma unroll
  for (int j = 0; j < NJ; ++j) { acc0[j] = (v8f){}; acc1[j] = (v8f){}; }

  const half_t* arow0 = A + (size_t)(tm0 * 16 + l15) * lda + 8 * hlf;
  const half_t* arow1 = A + (size_t)(tm1 * 16 + l15) * lda + 8 * hlf;

  for (int kc = 0; kc < Kp; kc += LDS_CHUNK) {
    const int clen = (Kp - kc < LDS_CHUNK) ? (Kp - kc) : LDS_CHUNK; // mult of 32
    // ---- cooperative stage of BT[n0 .. n0+NJ*16) x [kc .. kc+clen) --------
    const int c8n   = clen >> 3;            // 16B units per row
    const int units = NJ * 16 * c8n;
    for (int u = threadIdx.x; u < units; u += 256) {
      const int row = u / c8n;
      const int c8  = u - row * c8n;
      *(uint4*)(&lds[row * LDS_STRIDE + c8 * 8]) =
          *(const uint4*)(BT + (size_t)(n0 + row) * ldb + kc + c8 * 8);
    }
    __syncthreads();
    if (kc + LDS_CHUNK < Kp) {              // cover A latency across chunks
      __builtin_prefetch(arow0 + kc + LDS_CHUNK, 0, 1);
      __builtin_prefetch(arow1 + kc + LDS_CHUNK, 0, 1);
    }

    const int ksteps = clen >> 5;
    for (int ks = 0; ks < ksteps; ++ks) {
      FragU a0, a1;
      const uint4* pa0 = (const uint4*)(arow0 + kc + ks * 32);
      const uint4* pa1 = (const uint4*)(arow1 + kc + ks * 32);
      a0.u[0] = pa0[0]; a0.u[1] = pa0[2];   // +16 halves
      a1.u[0] = pa1[0]; a1.u[1] = pa1[2];
#pragma unroll
      for (int j = 0; j < NJ; ++j) {
        FragU bf;
        const uint4* pb =
            (const uint4*)(&lds[(j * 16 + l15) * LDS_STRIDE + ks * 32 + 8 * hlf]);
        bf.u[0] = pb[0];
        bf.u[1] = pb[2];
        acc0[j] = __builtin_amdgcn_wmma_f32_16x16x32_f16(
            false, a0.h, false, bf.h, (short)0, acc0[j], false, false);
        acc1[j] = __builtin_amdgcn_wmma_f32_16x16x32_f16(
            false, a1.h, false, bf.h, (short)0, acc1[j], false, false);
      }
    }
    __syncthreads();
  }

  if (!wr0) return;                         // barriers done; safe to exit
#pragma unroll
  for (int j = 0; j < NJ; ++j) {
    const int n = n0 + j * 16 + l15;
    const float bv = (bias != nullptr && n < Nvalid) ? bias[n] : 0.f;
#pragma unroll
    for (int r = 0; r < 8; ++r) {
      const int mm = tmb * 16 + r + 8 * hlf;
      float v = actf(acc0[j][r] + bv, act);
      if (resid) v += resid[(size_t)mm * ldr + n];
      if (Cf) Cf[(size_t)mm * ldcf + n] = v;
      if (Ch) Ch[(size_t)mm * ldch + n] = (half_t)v;
    }
    if (wr1) {
#pragma unroll
      for (int r = 0; r < 8; ++r) {
        const int mm = (tmb + 1) * 16 + r + 8 * hlf;
        float v = actf(acc1[j][r] + bv, act);
        if (resid) v += resid[(size_t)mm * ldr + n];
        if (Cf) Cf[(size_t)mm * ldcf + n] = v;
        if (Ch) Ch[(size_t)mm * ldch + n] = (half_t)v;
      }
    }
  }
}

// Convert W[K,N] f32 (rows roff..roff+Ksrc) into padded transposed f16 WT[Np,Kp]
__global__ void k_prep_wt(const float* __restrict__ W, int Nsrc, int roff, int Ksrc,
                          half_t* __restrict__ dst, int Np, int Kp)
{
  int i = blockIdx.x * blockDim.x + threadIdx.x;
  if (i >= Np * Kp) return;
  int n = i / Kp, k = i - n * Kp;
  float v = (n < Nsrc && k < Ksrc) ? W[(size_t)(roff + k) * Nsrc + n] : 0.f;
  dst[i] = (half_t)v;
}

__global__ void k_cvt_pad(const float* __restrict__ src, int lds_, int K,
                          half_t* __restrict__ dst, int Kp, int M)
{
  int i = blockIdx.x * blockDim.x + threadIdx.x;
  if (i >= M * Kp) return;
  int m = i / Kp, k = i - m * Kp;
  dst[i] = (half_t)((k < K) ? src[(size_t)m * lds_ + k] : 0.f);
}

__global__ void k_zero_f32(float* p, int n) {
  int i = blockIdx.x * blockDim.x + threadIdx.x;
  if (i < n) p[i] = 0.f;
}
__global__ void k_zero_f16(half_t* p, int n) {
  int i = blockIdx.x * blockDim.x + threadIdx.x;
  if (i < n) p[i] = (half_t)0.f;
}
__global__ void k_copy_f32(float* d, const float* s, int n) {
  int i = blockIdx.x * blockDim.x + threadIdx.x;
  if (i < n) d[i] = s[i];
}
__global__ void k_count_idx(const int* __restrict__ idx, float* cnt, int n) {
  int i = blockIdx.x * blockDim.x + threadIdx.x;
  if (i < n) atomicAdd(&cnt[idx[i]], 1.f);
}

__global__ void k_edge_geom(const float* __restrict__ x, const int* __restrict__ row,
                            const int* __restrict__ col, float* __restrict__ diff,
                            float* __restrict__ radial, int E)
{
  int e = blockIdx.x * blockDim.x + threadIdx.x;
  if (e >= E) return;
  int r = row[e], c = col[e];
  float dx = x[(size_t)r*3+0] - x[(size_t)c*3+0];
  float dy = x[(size_t)r*3+1] - x[(size_t)c*3+1];
  float dz = x[(size_t)r*3+2] - x[(size_t)c*3+2];
  diff[(size_t)e*3+0] = dx; diff[(size_t)e*3+1] = dy; diff[(size_t)e*3+2] = dz;
  radial[e] = dx*dx + dy*dy + dz*dz;
}

// m1 = silu(Ar[row] + Ac[col] + radial*wr + Ea)   (Ea holds e1 bias already)
__global__ __launch_bounds__(256) void k_combine(
    const half_t* __restrict__ Ar, const half_t* __restrict__ Ac,
    const half_t* Ea, const float* __restrict__ radial,
    const float* __restrict__ wr, const int* __restrict__ row,
    const int* __restrict__ col, half_t* out, int E)
{
  int t = blockIdx.x * 256 + threadIdx.x;
  int e = t >> 5; if (e >= E) return;
  int lane = t & 31; int c0 = lane * 4;
  int r = row[e], c = col[e];
  float rad = radial[e];
  H4 a, b, ea, o;
  a.u  = *(const uint2*)(Ar + (size_t)r*128 + c0);
  b.u  = *(const uint2*)(Ac + (size_t)c*128 + c0);
  ea.u = *(const uint2*)(Ea + (size_t)e*128 + c0);
#pragma unroll
  for (int j = 0; j < 4; ++j) {
    float v = (float)a.h[j] + (float)b.h[j] + rad * wr[c0+j] + (float)ea.h[j];
    o.h[j] = (half_t)(v / (1.f + __expf(-v)));
  }
  *(uint2*)(out + (size_t)e*128 + c0) = o.u;
}

// attention gate + segment_sum: m *= sigmoid(m . attw + attb); aggh[row] += m
__global__ __launch_bounds__(256) void k_gate_agg(
    half_t* m, const float* __restrict__ attw, const float* __restrict__ attb,
    const int* __restrict__ row, float* aggh, int E)
{
  int t = blockIdx.x * 256 + threadIdx.x;
  int e = t >> 5; if (e >= E) return;
  int lane = t & 31; int c0 = lane * 4;
  H4 mv; mv.u = *(const uint2*)(m + (size_t)e*128 + c0);
  float v0 = (float)mv.h[0], v1 = (float)mv.h[1], v2 = (float)mv.h[2], v3 = (float)mv.h[3];
  float s = v0*attw[c0] + v1*attw[c0+1] + v2*attw[c0+2] + v3*attw[c0+3];
  for (int o = 16; o > 0; o >>= 1) s += __shfl_xor(s, o, 32);
  float g = 1.f / (1.f + __expf(-(s + attb[0])));
  v0 *= g; v1 *= g; v2 *= g; v3 *= g;
  H4 ov; ov.h[0]=(half_t)v0; ov.h[1]=(half_t)v1; ov.h[2]=(half_t)v2; ov.h[3]=(half_t)v3;
  *(uint2*)(m + (size_t)e*128 + c0) = ov.u;
  float* ag = aggh + (size_t)row[e]*128 + c0;
  atomicAdd(ag+0, v0); atomicAdd(ag+1, v1); atomicAdd(ag+2, v2); atomicAdd(ag+3, v3);
}

// t = tanh(c1out . c2w); xagg[row] += diff * t
__global__ __launch_bounds__(256) void k_coord(
    const half_t* __restrict__ c1o, const float* __restrict__ c2w,
    const float* __restrict__ diff, const int* __restrict__ row,
    float* xagg, int E)
{
  int t = blockIdx.x * 256 + threadIdx.x;
  int e = t >> 5; if (e >= E) return;
  int lane = t & 31; int c0 = lane * 4;
  H4 cv; cv.u = *(const uint2*)(c1o + (size_t)e*128 + c0);
  float s = (float)cv.h[0]*c2w[c0] + (float)cv.h[1]*c2w[c0+1]
          + (float)cv.h[2]*c2w[c0+2] + (float)cv.h[3]*c2w[c0+3];
  for (int o = 16; o > 0; o >>= 1) s += __shfl_xor(s, o, 32);
  float tt = tanhf(s);
  if (lane < 3)
    atomicAdd(&xagg[(size_t)row[e]*3 + lane], diff[(size_t)e*3 + lane] * tt);
}

__global__ void k_xupdate(float* x, const float* __restrict__ xagg,
                          const float* __restrict__ cnt, int NN)
{
  int i = blockIdx.x * blockDim.x + threadIdx.x;
  if (i >= NN * 3) return;
  x[i] += xagg[i] / fmaxf(cnt[i / 3], 1.f);
}

__global__ void k_build_xn(const half_t* __restrict__ h16, const float* __restrict__ aggh,
                           half_t* __restrict__ xn, int NN)
{
  int i = blockIdx.x * blockDim.x + threadIdx.x;
  if (i >= NN * 128) return;
  int n = i >> 7, c = i & 127;
  xn[(size_t)n*256 + c]       = h16[i];
  xn[(size_t)n*256 + 128 + c] = (half_t)aggh[i];
}

__global__ void k_segsum(const float* __restrict__ h, int ldh, int F,
                         const int* __restrict__ batch, float* out, int NN)
{
  int i = blockIdx.x * blockDim.x + threadIdx.x;
  if (i >= NN * F) return;
  int n = i / F, f = i - n * F;
  atomicAdd(&out[(size_t)batch[n]*F + f], h[(size_t)n*ldh + f]);
}
__global__ void k_divrows(float* s, const float* __restrict__ cnt, int F, int n) {
  int i = blockIdx.x * blockDim.x + threadIdx.x;
  if (i < n) s[i] /= fmaxf(cnt[i / F], 1.f);
}

// training-mode BatchNorm1d over `rows` rows + activation
__global__ void k_bn_act(const float* __restrict__ in, int ldin, int F,
                         const float* __restrict__ g, const float* __restrict__ b,
                         int act, int rows,
                         half_t* o16, int ldo, float* of, int ldof)
{
  int f = blockIdx.x * blockDim.x + threadIdx.x;
  if (f >= F) return;
  float mu = 0.f, sq = 0.f;
  for (int r = 0; r < rows; ++r) {
    float v = in[(size_t)r*ldin + f]; mu += v; sq += v*v;
  }
  float inv = 1.f / rows;
  mu *= inv;
  float var = sq * inv - mu * mu;
  float sc = g[f] * rsqrtf(var + 1e-5f);
  float sh = b[f] - mu * sc;
  for (int r = 0; r < rows; ++r) {
    float v = actf(sc * in[(size_t)r*ldin + f] + sh, act);
    if (o16) o16[(size_t)r*ldo + f] = (half_t)v;
    if (of)  of[(size_t)r*ldof + f] = v;
  }
}

__global__ void k_add_tanh(const float* a, const float* b, int n,
                           half_t* o16, float* of)
{
  int i = blockIdx.x * blockDim.x + threadIdx.x;
  if (i >= n) return;
  float v = tanhf(a[i] + b[i]);
  o16[i] = (half_t)v;
  of[i]  = v;
}

__global__ void k_copy_out(float* out, const float* src) {
  int i = threadIdx.x;
  if (i < 64) out[i] = src[(i >> 1) * 16 + (i & 1)];
}

// ---------------------------------------------------------------------------
extern "C" void kernel_launch(void* const* d_in, const int* in_sizes, int n_in,
                              void* d_out, int out_size, void* d_ws, size_t ws_size,
                              hipStream_t stream)
{
  (void)out_size; (void)ws_size;
  const float* x_res     = (const float*)d_in[0];
  const float* x_pos     = (const float*)d_in[1];
  const float* edge_attr = (const float*)d_in[2];
  const float* x_form    = (const float*)d_in[3];
  const float* x_AA      = (const float*)d_in[4];
  const int*   edge_idx  = (const int*)d_in[n_in - 2];
  const int*   x_batch   = (const int*)d_in[n_in - 1];
  const int NN = in_sizes[0] / 1280;
  const int E  = in_sizes[2] / 32;
  const int B  = in_sizes[3] / 2048;
  const int* rowi = edge_idx;
  const int* coli = edge_idx + E;

  // --- parameter walker (reference dict insertion order) -------------------
  int pi = 5;
  auto nxt = [&]() -> const float* { return (const float*)d_in[pi++]; };
  struct Lin { const float* w; const float* b; };
  auto nlin = [&]() -> Lin { Lin l; l.w = nxt(); l.b = nxt(); return l; };
  struct Egcl { Lin e1, e2, att, c1; const float* c2w; Lin n1, n2; };
  struct Egnn { Lin ein, eout; Egcl l[4]; };
  auto negnn = [&]() -> Egnn {
    Egnn g; g.ein = nlin(); g.eout = nlin();
    for (int i = 0; i < 4; ++i) {
      Egcl c; c.e1 = nlin(); c.e2 = nlin(); c.att = nlin(); c.c1 = nlin();
      c.c2w = nxt(); c.n1 = nlin(); c.n2 = nlin(); g.l[i] = c;
    }
    return g;
  };
  Egnn eg1 = negnn(), eg2 = negnn(), eg3 = negnn();
  Lin bnr1 = nlin(), bnr2 = nlin(), bnr3 = nlin();
  Lin fcf[4]; for (int i = 0; i < 4; ++i) fcf[i] = nlin();
  Lin bnf[4]; for (int i = 0; i < 4; ++i) bnf[i] = nlin();
  Lin fca[4]; for (int i = 0; i < 4; ++i) fca[i] = nlin();
  Lin bna[4]; for (int i = 0; i < 4; ++i) bna[i] = nlin();
  struct RB { Lin fc1, bn1, fc2, bn2, sc; };
  RB rb[4];
  rb[0].fc1 = nlin(); rb[0].bn1 = nlin(); rb[0].fc2 = nlin(); rb[0].bn2 = nlin(); rb[0].sc = nlin();
  for (int i = 1; i < 4; ++i) { rb[i].fc1 = nlin(); rb[i].bn1 = nlin(); rb[i].fc2 = nlin(); rb[i].bn2 = nlin(); }
  Lin fin = nlin();

  // --- workspace bump allocator -------------------------------------------
  char* base = (char*)d_ws; size_t off = 0;
  auto alloc = [&](size_t bytes) -> char* {
    char* p = base + off; off += (bytes + 255) & ~(size_t)255; return p;
  };
  float*  x      = (float*) alloc((size_t)NN * 3 * 4);
  float*  cntE   = (float*) alloc((size_t)NN * 4);
  float*  cntB   = (float*) alloc((size_t)B * 4);
  float*  diff   = (float*) alloc((size_t)E * 3 * 4);
  float*  radial = (float*) alloc((size_t)E * 4);
  half_t* ea16   = (half_t*)alloc((size_t)E * 32 * 2);
  half_t* bufE1  = (half_t*)alloc((size_t)E * 128 * 2);
  half_t* bufE2  = (half_t*)alloc((size_t)E * 128 * 2);
  float*  aggh   = (float*) alloc((size_t)NN * 128 * 4);
  float*  xagg   = (float*) alloc((size_t)NN * 3 * 4);
  float*  hf     = (float*) alloc((size_t)NN * 128 * 4);
  half_t* h16    = (half_t*)alloc((size_t)NN * 128 * 2);
  half_t* Ar16   = (half_t*)alloc((size_t)NN * 128 * 2);
  half_t* Ac16   = (half_t*)alloc((size_t)NN * 128 * 2);
  half_t* Xn16   = (half_t*)alloc((size_t)NN * 256 * 2);
  half_t* n1o16  = (half_t*)alloc((size_t)NN * 128 * 2);
  half_t* Ain16  = (half_t*)alloc((size_t)NN * 1280 * 2);
  float*  h1f    = (float*) alloc((size_t)NN * 160 * 4);
  float*  h2f    = (float*) alloc((size_t)NN * 80 * 4);
  float*  h3f    = (float*) alloc((size_t)NN * 48 * 4);
  half_t* wt     = (half_t*)alloc((size_t)320 * 2048 * 2);   // max WT
  float*  segf   = (float*) alloc((size_t)B * 160 * 4);
  float*  smCf   = (float*) alloc((size_t)B * 320 * 4);
  float*  rbTf   = (float*) alloc((size_t)B * 128 * 4);
  float*  rbG    = (float*) alloc((size_t)B * 32 * 4);
  float*  rbV    = (float*) alloc((size_t)B * 32 * 4);
  float*  rbS    = (float*) alloc((size_t)B * 32 * 4);
  float*  rbXf   = (float*) alloc((size_t)B * 32 * 4);
  float*  outst  = (float*) alloc((size_t)B * 16 * 4);
  // small f16 pool (zeroed once -> pad columns stay zero)
  half_t* smpool = (half_t*)alloc((size_t)B * 3552 * 2);
  size_t sp = 0;
  auto salloc = [&](int ld) -> half_t* { half_t* p = smpool + sp; sp += (size_t)B * ld; return p; };
  half_t* smA0 = salloc(2048); half_t* smB1 = salloc(320);
  half_t* smB2 = salloc(160);  half_t* smB3 = salloc(96);
  half_t* saA0 = salloc(32);   half_t* saB1 = salloc(160);
  half_t* saB2 = salloc(128);  half_t* saB3 = salloc(32);
  half_t* resin = salloc(416); half_t* rb16 = salloc(128);
  half_t* rbX16 = salloc(32);

  auto cdiv = [](long a, long b) { return (int)((a + b - 1) / b); };
  auto zf = [&](float* p, int n) { k_zero_f32<<<cdiv(n,256),256,0,stream>>>(p, n); };

  // GEMM launcher: prepares WT f16 then runs WMMA kernel(s).
  // Column space split into full 64-wide blocks (NJ=4) + one remainder launch.
  // Each block covers 256 rows (8 waves x 2 M-tiles).
  auto gemm = [&](const half_t* A, int lda, const float* W, int Ksrc, int roff, int Nsrc,
                  int M, int Kp, int Np, const float* bias, int Nvalid, int act,
                  const float* resid, int ldr, float* Cf, int ldcf, half_t* Ch, int ldch) {
    int tot = Np * Kp;
    k_prep_wt<<<cdiv(tot,256),256,0,stream>>>(W, Nsrc, roff, Ksrc, wt, Np, Kp);
    int Mt = M / 16;
    int gx = cdiv(Mt, 16);
    int full = Np / 64;
    int rem  = (Np - full * 64) / 16;
    if (full)
      k_wmma_gemm<4><<<dim3(gx,full),256,0,stream>>>(A,lda,wt,Kp,Mt,0,Kp,
          bias,Nvalid,act,resid,ldr,Cf,ldcf,Ch,ldch);
    int n0b = full * 64;
    if (rem == 1)
      k_wmma_gemm<1><<<dim3(gx,1),256,0,stream>>>(A,lda,wt,Kp,Mt,n0b,Kp,
          bias,Nvalid,act,resid,ldr,Cf,ldcf,Ch,ldch);
    else if (rem == 2)
      k_wmma_gemm<2><<<dim3(gx,1),256,0,stream>>>(A,lda,wt,Kp,Mt,n0b,Kp,
          bias,Nvalid,act,resid,ldr,Cf,ldcf,Ch,ldch);
    else if (rem == 3)
      k_wmma_gemm<3><<<dim3(gx,1),256,0,stream>>>(A,lda,wt,Kp,Mt,n0b,Kp,
          bias,Nvalid,act,resid,ldr,Cf,ldcf,Ch,ldch);
  };

  // --- one-time prep per call ---------------------------------------------
  k_copy_f32<<<cdiv(NN*3,256),256,0,stream>>>(x, x_pos, NN*3);
  zf(cntE, NN); zf(cntB, B);
  k_count_idx<<<cdiv(E,256),256,0,stream>>>(rowi, cntE, E);
  k_count_idx<<<cdiv(NN,256),256,0,stream>>>(x_batch, cntB, NN);
  k_cvt_pad<<<cdiv((long)E*32,256),256,0,stream>>>(edge_attr, 32, 32, ea16, 32, E);
  k_zero_f16<<<cdiv((long)sp,256),256,0,stream>>>(smpool, (int)sp);

  // --- one E_GCL layer -----------------------------------------------------
  auto egcl = [&](const Egcl& L) {
    k_edge_geom<<<cdiv(E,256),256,0,stream>>>(x, rowi, coli, diff, radial, E);
    // per-node halves of e1:  Ar = h@W[0:128],  Ac = h@W[128:256]
    gemm(h16,128, L.e1.w,128,0,  128, NN,128,128, nullptr,0,ACT_NONE, nullptr,0, nullptr,0, Ar16,128);
    gemm(h16,128, L.e1.w,128,128,128, NN,128,128, nullptr,0,ACT_NONE, nullptr,0, nullptr,0, Ac16,128);
    // edge-attr part (+ bias):  Ea = edge_attr@W[257:289] + b
    gemm(ea16,32, L.e1.w,32,257, 128, E, 32, 128, L.e1.b,128,ACT_NONE, nullptr,0, nullptr,0, bufE1,128);
    // m1 = silu(Ar[row] + Ac[col] + radial*W[256] + Ea)   (in place in bufE1)
    k_combine<<<cdiv((long)E*32,256),256,0,stream>>>(Ar16, Ac16, bufE1, radial,
                                                     L.e1.w + 256*128, rowi, coli, bufE1, E);
    // m2 = silu(m1@e2 + b)
    gemm(bufE1,128, L.e2.w,128,0,128, E,128,128, L.e2.b,128,ACT_SILU, nullptr,0, nullptr,0, bufE2,128);
    zf(aggh, NN*128);
    k_gate_agg<<<cdiv((long)E*32,256),256,0,stream>>>(bufE2, L.att.w, L.att.b, rowi, aggh, E);
    // coord branch: c1 then tanh(dot c2w), mean-aggregate
    gemm(bufE2,128, L.c1.w,128,0,128, E,128,128, L.c1.b,128,ACT_SILU, nullptr,0, nullptr,0, bufE1,128);
    zf(xagg, NN*3);
    k_coord<<<cdiv((long)E*32,256),256,0,stream>>>(bufE1, L.c2w, diff, rowi, xagg, E);
    k_xupdate<<<cdiv(NN*3,256),256,0,stream>>>(x, xagg, cntE, NN);
    // node branch: h = h + n2(silu(n1([h|aggh])))
    k_build_xn<<<cdiv(NN*128,256),256,0,stream>>>(h16, aggh, Xn16, NN);
    gemm(Xn16,256, L.n1.w,256,0,128, NN,256,128, L.n1.b,128,ACT_SILU, nullptr,0, nullptr,0, n1o16,128);
    gemm(n1o16,128, L.n2.w,128,0,128, NN,128,128, L.n2.b,128,ACT_NONE, hf,128, hf,128, h16,128);
  };

  auto run_egnn = [&](const Egnn& G, const half_t* Ain, int KpIn, int in_nf,
                      int out_nf, int NpOut, float* hout, int ldho) {
    gemm(Ain,KpIn, G.ein.w,in_nf,0,128, NN,KpIn,128, G.ein.b,128,ACT_NONE,
         nullptr,0, hf,128, h16,128);
    for (int i = 0; i < 4; ++i) egcl(G.l[i]);
    gemm(h16,128, G.eout.w,128,0,out_nf, NN,128,NpOut, G.eout.b,out_nf,ACT_NONE,
         nullptr,0, hout,ldho, nullptr,0);
  };

  // seg_mean over graphs + BN/ReLU into resin at column offset
  auto segbn = [&](const float* hsrc, int ldh, int F, const Lin& bnp, int cof) {
    zf(segf, B*F);
    k_segsum<<<cdiv((long)NN*F,256),256,0,stream>>>(hsrc, ldh, F, x_batch, segf, NN);
    k_divrows<<<cdiv(B*F,256),256,0,stream>>>(segf, cntB, F, B*F);
    k_bn_act<<<cdiv(F,256),256,0,stream>>>(segf, F, F, bnp.w, bnp.b, ACT_RELU, B,
                                           resin + cof, 416, nullptr, 0);
  };

  // --- EGNN stacks ---------------------------------------------------------
  k_cvt_pad<<<cdiv((long)NN*1280,256),256,0,stream>>>(x_res, 1280, 1280, Ain16, 1280, NN);
  run_egnn(eg1, Ain16, 1280, 1280, 160, 160, h1f, 160);
  segbn(h1f, 160, 160, bnr1, 0);
  k_cvt_pad<<<cdiv((long)NN*160,256),256,0,stream>>>(h1f, 160, 160, Ain16, 160, NN);
  run_egnn(eg2, Ain16, 160, 160, 80, 80, h2f, 80);
  segbn(h2f, 80, 80, bnr2, 270);
  k_cvt_pad<<<cdiv((long)NN*96,256),256,0,stream>>>(h2f, 80, 80, Ain16, 96, NN);
  run_egnn(eg3, Ain16, 96, 80, 40, 48, h3f, 48);
  segbn(h3f, 48, 40, bnr3, 350);

  // --- formula MLP (fcf/bnf) -> resin[160:240] -----------------------------
  k_cvt_pad<<<cdiv((long)B*2048,256),256,0,stream>>>(x_form, 2048, 2048, smA0, 2048, B);
  gemm(smA0,2048, fcf[0].w,2048,0,320, B,2048,320, fcf[0].b,320,ACT_NONE, nullptr,0, smCf,320, nullptr,0);
  k_bn_act<<<cdiv(320,256),256,0,stream>>>(smCf,320,320, bnf[0].w,bnf[0].b, ACT_RELU,B, smB1,320, nullptr,0);
  gemm(smB1,320, fcf[1].w,320,0,160, B,320,160, fcf[1].b,160,ACT_NONE, nullptr,0, smCf,160, nullptr,0);
  k_bn_act<<<1,256,0,stream>>>(smCf,160,160, bnf[1].w,bnf[1].b, ACT_RELU,B, smB2,160, nullptr,0);
  gemm(smB2,160, fcf[2].w,160,0,80, B,160,80, fcf[2].b,80,ACT_NONE, nullptr,0, smCf,80, nullptr,0);
  k_bn_act<<<1,256,0,stream>>>(smCf,80,80, bnf[2].w,bnf[2].b, ACT_RELU,B, smB3,96, nullptr,0);
  gemm(smB3,96, fcf[3].w,80,0,80, B,96,80, fcf[3].b,80,ACT_NONE, nullptr,0, smCf,80, nullptr,0);
  k_bn_act<<<1,256,0,stream>>>(smCf,80,80, bnf[3].w,bnf[3].b, ACT_RELU,B, resin+160,416, nullptr,0);

  // --- AA MLP (fca/bna) -> resin[240:270] ----------------------------------
  k_cvt_pad<<<cdiv((long)B*32,256),256,0,stream>>>(x_AA, 20, 20, saA0, 32, B);
  gemm(saA0,32, fca[0].w,20,0,160, B,32,160, fca[0].b,160,ACT_NONE, nullptr,0, smCf,160, nullptr,0);
  k_bn_act<<<1,256,0,stream>>>(smCf,160,160, bna[0].w,bna[0].b, ACT_RELU,B, saB1,160, nullptr,0);
  gemm(saB1,160, fca[1].w,160,0,120, B,160,128, fca[1].b,120,ACT_NONE, nullptr,0, smCf,128, nullptr,0);
  k_bn_act<<<1,256,0,stream>>>(smCf,128,120, bna[1].w,bna[1].b, ACT_RELU,B, saB2,128, nullptr,0);
  gemm(saB2,128, fca[2].w,120,0,30, B,128,32, fca[2].b,30,ACT_NONE, nullptr,0, smCf,32, nullptr,0);
  k_bn_act<<<1,256,0,stream>>>(smCf,32,30, bna[2].w,bna[2].b, ACT_RELU,B, saB3,32, nullptr,0);
  gemm(saB3,32, fca[3].w,30,0,30, B,32,32, fca[3].b,30,ACT_NONE, nullptr,0, smCf,32, nullptr,0);
  k_bn_act<<<1,256,0,stream>>>(smCf,32,30, bna[3].w,bna[3].b, ACT_RELU,B, resin+240,416, nullptr,0);

  // --- residual blocks over resin[32,390->416] -----------------------------
  gemm(resin,416, rb[0].fc1.w,390,0,125, B,416,128, rb[0].fc1.b,125,ACT_NONE, nullptr,0, rbTf,128, nullptr,0);
  k_bn_act<<<1,256,0,stream>>>(rbTf,128,125, rb[0].bn1.w,rb[0].bn1.b, ACT_TANH,B, rb16,128, nullptr,0);
  gemm(rb16,128, rb[0].fc2.w,125,0,32, B,128,32, rb[0].fc2.b,32,ACT_NONE, nullptr,0, rbG,32, nullptr,0);
  k_bn_act<<<1,256,0,stream>>>(rbG,32,32, rb[0].bn2.w,rb[0].bn2.b, ACT_NONE,B, nullptr,0, rbV,32);
  gemm(resin,416, rb[0].sc.w,390,0,32, B,416,32, rb[0].sc.b,32,ACT_NONE, nullptr,0, rbS,32, nullptr,0);
  k_add_tanh<<<cdiv(B*32,256),256,0,stream>>>(rbV, rbS, B*32, rbX16, rbXf);
  for (int i = 1; i < 4; ++i) {
    gemm(rbX16,32, rb[i].fc1.w,32,0,32, B,32,32, rb[i].fc1.b,32,ACT_NONE, nullptr,0, rbTf,32, nullptr,0);
    k_bn_act<<<1,256,0,stream>>>(rbTf,32,32, rb[i].bn1.w,rb[i].bn1.b, ACT_TANH,B, rb16,32, nullptr,0);
    gemm(rb16,32, rb[i].fc2.w,32,0,32, B,32,32, rb[i].fc2.b,32,ACT_NONE, nullptr,0, rbG,32, nullptr,0);
    k_bn_act<<<1,256,0,stream>>>(rbG,32,32, rb[i].bn2.w,rb[i].bn2.b, ACT_NONE,B, nullptr,0, rbV,32);
    k_add_tanh<<<cdiv(B*32,256),256,0,stream>>>(rbV, rbXf, B*32, rbX16, rbXf);
  }

  // --- final classifier ----------------------------------------------------
  gemm(rbX16,32, fin.w,32,0,2, B,32,16, fin.b,2,ACT_NONE, nullptr,0, outst,16, nullptr,0);
  k_copy_out<<<1,64,0,stream>>>((float*)d_out, outst);
}
